// ProgressivePruningSystem_69569880261170
// MI455X (gfx1250) — matrix operand order repaired and checked by reading, not compiled, not run on hardware
//
#include <hip/hip_runtime.h>
#include <hip/hip_bf16.h>

// ---------------------------------------------------------------------------
// ProgressivePruningSystem on MI455X (gfx1250, wave32)
//   k0: f32 -> bf16 casts (x, w1, w2)
//   k1: h = gelu(x @ w1^T + b1)      WMMA bf16, async-to-LDS double buffered
//   k2: logits = h @ w2^T + b2 ; full routing epilogue -> probs (token,80)
//   k3: out = sum_p path[p] * probs  (bandwidth-bound streaming, float4)
// Workspace: xb(33.5MB) w1b(16.8MB) w2b(0.63MB) hb(67MB) probs(2.6MB) ~121MB
// ---------------------------------------------------------------------------

typedef __attribute__((ext_vector_type(16))) __bf16 v16bf;
typedef __attribute__((ext_vector_type(8)))  __bf16 v8bf;
typedef __attribute__((ext_vector_type(8)))  float  v8f;
typedef __attribute__((ext_vector_type(4)))  float  v4f;

#define TOKENS   8192       // B*L
#define HID      2048
#define HID2     4096
#define NHEAD    16
#define NPATH    5
#define HD       128
#define NOUT     80         // NHEAD*NPATH

#define EPS_MAX_F  (0.05f * (1.0f - 2500.0f/3000.0f))
#define THRESH_F   0.00025f

#if __has_builtin(__builtin_amdgcn_s_wait_asynccnt)
#define WAIT_ASYNC(n) __builtin_amdgcn_s_wait_asynccnt(n)
#else
#define WAIT_ASYNC(n) asm volatile("s_wait_asynccnt %0" :: "i"(n) : "memory")
#endif

// One b128 async copy per lane: LDS[lds_off] <- MEM[gaddr], tracked on ASYNCcnt.
__device__ __forceinline__ void async_g2l_b128(unsigned lds_off, const void* gaddr) {
    asm volatile("global_load_async_to_lds_b128 %0, %1, off"
                 :: "v"(lds_off), "v"(gaddr) : "memory");
}

__device__ __forceinline__ v16bf cat16(v8bf lo, v8bf hi) {
    return __builtin_shufflevector(lo, hi, 0,1,2,3,4,5,6,7,8,9,10,11,12,13,14,15);
}

// ---------------------------------------------------------------- k0: casts
__global__ void cast_f32_to_bf16(const float* __restrict__ in,
                                 __bf16* __restrict__ out, int n) {
    int i = blockIdx.x * blockDim.x + threadIdx.x;
    int stride = gridDim.x * blockDim.x;
    for (; i < n; i += stride) out[i] = (__bf16)in[i];
}

// ------------------------------------------------- k1: GEMM1 (M8192 N4096 K2048)
// block tile 128x128, BK=32, 8 waves, double-buffered async-to-LDS staging.
#define LPITCH 80           // bytes per LDS row: 32 bf16 (64B) + 16B pad
#define ABUF_SZ (128 * LPITCH)           // 10240 B
#define TILE_SZ (2 * ABUF_SZ)            // A + B per buffer

__device__ __forceinline__ void stage_tiles(const __bf16* __restrict__ A,
                                            const __bf16* __restrict__ B,
                                            int rowA0, int rowB0, int k0,
                                            unsigned lbase, int tid) {
#pragma unroll
    for (int c = 0; c < 2; ++c) {                 // 2 A + 2 B = 4 async ops/lane
        int chunk = tid + c * 256;                // 512 chunks of 16B per 8KB tile
        int r = chunk >> 2;
        int s = chunk & 3;
        unsigned loff = lbase + (unsigned)(r * LPITCH + s * 16);
        async_g2l_b128(loff,           A + (size_t)(rowA0 + r) * HID + k0 + s * 8);
        async_g2l_b128(loff + ABUF_SZ, B + (size_t)(rowB0 + r) * HID + k0 + s * 8);
    }
}

__global__ __launch_bounds__(256, 1)
void gemm1_gelu_kernel(const __bf16* __restrict__ A,   // x_bf16 [8192][2048]
                       const __bf16* __restrict__ B,   // w1_bf16 [4096][2048]
                       const float*  __restrict__ b1,  // [4096]
                       __bf16* __restrict__ H) {       // out [8192][4096]
    __shared__ __align__(16) unsigned char smem[2 * TILE_SZ];
    const unsigned smem_base = (unsigned)(uintptr_t)smem;

    const int tid  = threadIdx.x;
    const int lane = tid & 31;
    const int wid  = tid >> 5;
    const int lm   = lane & 15;
    const int lh   = lane >> 4;           // lane half
    const int wm   = wid & 3;             // 4 waves x 32 rows
    const int wn   = wid >> 2;            // 2 waves x 64 cols
    const int rowA0 = blockIdx.y * 128;
    const int rowB0 = blockIdx.x * 128;

    v8f acc[2][4];
#pragma unroll
    for (int i = 0; i < 2; ++i)
#pragma unroll
        for (int j = 0; j < 4; ++j) acc[i][j] = (v8f)0.0f;

    constexpr int KTILES = HID / 32;      // 64
    stage_tiles(A, B, rowA0, rowB0, 0, smem_base, tid);

    for (int kt = 0; kt < KTILES; ++kt) {
        const int buf = kt & 1;
        if (kt + 1 < KTILES) {
            stage_tiles(A, B, rowA0, rowB0, (kt + 1) * 32,
                        smem_base + (buf ^ 1) * TILE_SZ, tid);
            WAIT_ASYNC(4);                // current tile's 4 ops retired (in-order)
        } else {
            WAIT_ASYNC(0);
        }
        __syncthreads();

        const unsigned char* Ab = smem + buf * TILE_SZ;
        const unsigned char* Bb = Ab + ABUF_SZ;

        // A frag: elems 0..7 = K(lh*8..), elems 8..15 = K(16+lh*8..)
        v16bf af[2];
#pragma unroll
        for (int jm = 0; jm < 2; ++jm) {
            const unsigned char* p = Ab + (wm * 32 + jm * 16 + lm) * LPITCH + lh * 16;
            af[jm] = cat16(*(const v8bf*)p, *(const v8bf*)(p + 32));
        }
        // B frag: elems 0..7 = K(lh*8..), elems 8..15 = K(16+lh*8..)
        v16bf bf[4];
#pragma unroll
        for (int jn = 0; jn < 4; ++jn) {
            const unsigned char* p = Bb + (wn * 64 + jn * 16 + lm) * LPITCH + lh * 16;
            bf[jn] = cat16(*(const v8bf*)p, *(const v8bf*)(p + 32));
        }
#pragma unroll
        for (int jm = 0; jm < 2; ++jm)
#pragma unroll
            for (int jn = 0; jn < 4; ++jn)
                acc[jm][jn] = __builtin_amdgcn_wmma_f32_16x16x32_bf16(
                    false, af[jm], false, bf[jn], (short)0, acc[jm][jn], false, false);
        __syncthreads();                  // before next iteration overwrites buf
    }

    // epilogue: bias + exact GELU, store bf16
#pragma unroll
    for (int jm = 0; jm < 2; ++jm)
#pragma unroll
        for (int jn = 0; jn < 4; ++jn) {
            const int col = rowB0 + wn * 64 + jn * 16 + lm;
            const float bias = b1[col];
#pragma unroll
            for (int j = 0; j < 8; ++j) {
                const int row = rowA0 + wm * 32 + jm * 16 + j + lh * 8;
                float v = acc[jm][jn][j] + bias;
                float g = 0.5f * v * (1.0f + erff(v * 0.70710678118654752f));
                H[(size_t)row * HID2 + col] = (__bf16)g;
            }
        }
}

// ---------------------------------- k2: logits GEMM (N=80) + routing epilogue
__global__ __launch_bounds__(32, 1)
void router_kernel(const __bf16* __restrict__ Hb,       // [8192][4096]
                   const __bf16* __restrict__ W2,       // [80][4096]
                   const float*  __restrict__ b2,       // [80]
                   const float*  __restrict__ log_temp, // [16]
                   const float*  __restrict__ eps_logit,// [16][5]
                   float* __restrict__ probs_out) {     // [8192][80]
    __shared__ float lg[16 * NOUT];
    const int lane = threadIdx.x;
    const int lm = lane & 15;
    const int lh = lane >> 4;
    const int m0 = blockIdx.x * 16;

    v8f acc[NPATH];
#pragma unroll
    for (int t = 0; t < NPATH; ++t) acc[t] = (v8f)0.0f;

    for (int kt = 0; kt < HID2 / 32; ++kt) {
        const int k0 = kt * 32;
        const __bf16* pa = Hb + (size_t)(m0 + lm) * HID2 + k0 + lh * 8;
        v16bf a = cat16(*(const v8bf*)pa, *(const v8bf*)(pa + 16));
#pragma unroll
        for (int t = 0; t < NPATH; ++t) {
            const __bf16* pb = W2 + (size_t)(t * 16 + lm) * HID2 + k0 + lh * 8;
            v16bf b = cat16(*(const v8bf*)pb, *(const v8bf*)(pb + 16));
            acc[t] = __builtin_amdgcn_wmma_f32_16x16x32_bf16(
                false, a, false, b, (short)0, acc[t], false, false);
        }
    }

    // scatter logits (bias + temperature) to LDS: lg[token][n]
#pragma unroll
    for (int t = 0; t < NPATH; ++t) {
        const int n = t * 16 + lm;
        const float inv_t = __expf(-log_temp[n / NPATH]);
#pragma unroll
        for (int j = 0; j < 8; ++j) {
            const int row = j + lh * 8;
            lg[row * NOUT + n] = (acc[t][j] + b2[n]) * inv_t;
        }
    }
    __syncthreads();

    // per-(token,head) routing math: 256 pairs over 32 lanes
#pragma unroll
    for (int i = 0; i < 8; ++i) {
        const int pair = lane + 32 * i;
        const int tok  = pair >> 4;
        const int head = pair & 15;
        float pr[NPATH], orig[NPATH], ev[NPATH];

        float mx = -1e30f;
#pragma unroll
        for (int p = 0; p < NPATH; ++p) {
            pr[p] = lg[tok * NOUT + head * NPATH + p];
            mx = fmaxf(mx, pr[p]);
        }
        float s = 0.0f;
#pragma unroll
        for (int p = 0; p < NPATH; ++p) { pr[p] = __expf(pr[p] - mx); s += pr[p]; }
        const float inv = 1.0f / s;
        float pmax = 0.0f;
#pragma unroll
        for (int p = 0; p < NPATH; ++p) { pr[p] *= inv; pmax = fmaxf(pmax, pr[p]); }

        float esum = 0.0f;
#pragma unroll
        for (int p = 0; p < NPATH; ++p) {
            const float sg = 1.0f / (1.0f + __expf(-eps_logit[head * NPATH + p]));
            ev[p] = EPS_MAX_F * (1.0f - pmax) * sg;
            esum += ev[p];
        }
#pragma unroll
        for (int p = 0; p < NPATH; ++p) {
            pr[p] = pr[p] * (1.0f - esum) + ev[p];
            pr[p] = fminf(fmaxf(pr[p], 1e-9f), 1.0f);
            orig[p] = pr[p];
        }
        // hard prune + renorm
        s = 0.0f;
#pragma unroll
        for (int p = 0; p < NPATH; ++p) { pr[p] = (pr[p] > THRESH_F) ? pr[p] : 0.0f; s += pr[p]; }
        s = fmaxf(s, 1e-9f);
        int cnt = 0;
#pragma unroll
        for (int p = 0; p < NPATH; ++p) { pr[p] /= s; cnt += (pr[p] > 1e-6f) ? 1 : 0; }
        // min-active via top-2 of orig (ties -> lower index, like lax.top_k)
        if (cnt < 2) {
            int i1 = 0; float v1 = orig[0];
#pragma unroll
            for (int p = 1; p < NPATH; ++p) if (orig[p] > v1) { v1 = orig[p]; i1 = p; }
            int i2 = -1; float v2 = -1.0f;
#pragma unroll
            for (int p = 0; p < NPATH; ++p)
                if (p != i1 && orig[p] > v2) { v2 = orig[p]; i2 = p; }
#pragma unroll
            for (int p = 0; p < NPATH; ++p) pr[p] = (p == i1 || p == i2) ? 0.5f : 0.0f;
        }
        s = 0.0f;
#pragma unroll
        for (int p = 0; p < NPATH; ++p) s += pr[p];
        s = fmaxf(s, 1e-9f);
        float* o = probs_out + (size_t)(m0 + tok) * NOUT + head * NPATH;
#pragma unroll
        for (int p = 0; p < NPATH; ++p)
            o[p] = fminf(fmaxf(pr[p] / s, 1e-9f), 1.0f);
    }
}

// --------------------------------------- k3: combine paths (bandwidth bound)
__global__ void combine_kernel(const float* __restrict__ paths,  // [5][8192][16][128]
                               const float* __restrict__ probs,  // [8192][80]
                               float* __restrict__ out) {        // [8192][16][128]
    const int idx = blockIdx.x * blockDim.x + threadIdx.x;       // float4 index
    constexpr int TOTAL4 = TOKENS * NHEAD * HD / 4;
    if (idx >= TOTAL4) return;
    const int token = idx / (NHEAD * HD / 4);
    const int rem   = idx % (NHEAD * HD / 4);
    const int head  = rem / (HD / 4);
    const int d4    = rem % (HD / 4);
    const float* pr = probs + (size_t)token * NOUT + head * NPATH;
    const size_t off = ((size_t)token * NHEAD + head) * HD + d4 * 4;

    v4f accv = (v4f)0.0f;
#pragma unroll
    for (int p = 0; p < NPATH; ++p) {
        v4f v = *(const v4f*)(paths + (size_t)p * TOKENS * NHEAD * HD + off);
        accv += v * pr[p];
    }
    *(v4f*)(out + off) = accv;
}

// ---------------------------------------------------------------------------
extern "C" void kernel_launch(void* const* d_in, const int* in_sizes, int n_in,
                              void* d_out, int out_size, void* d_ws, size_t ws_size,
                              hipStream_t stream) {
    (void)in_sizes; (void)n_in; (void)out_size; (void)ws_size;
    const float* x         = (const float*)d_in[0];
    const float* w1        = (const float*)d_in[1];
    const float* b1        = (const float*)d_in[2];
    const float* w2        = (const float*)d_in[3];
    const float* b2        = (const float*)d_in[4];
    const float* log_temp  = (const float*)d_in[5];
    const float* eps_logit = (const float*)d_in[6];
    const float* paths     = (const float*)d_in[7];
    float* out = (float*)d_out;

    char* ws = (char*)d_ws;
    __bf16* xb  = (__bf16*)ws;  ws += (size_t)TOKENS * HID  * 2;   // 33.5 MB
    __bf16* w1b = (__bf16*)ws;  ws += (size_t)HID2   * HID  * 2;   // 16.8 MB
    __bf16* w2b = (__bf16*)ws;  ws += (size_t)NOUT   * HID2 * 2;   // 0.63 MB
    __bf16* hb  = (__bf16*)ws;  ws += (size_t)TOKENS * HID2 * 2;   // 67 MB
    float*  probs = (float*)ws;                                    // 2.6 MB

    cast_f32_to_bf16<<<2048, 256, 0, stream>>>(x,  xb,  TOKENS * HID);
    cast_f32_to_bf16<<<2048, 256, 0, stream>>>(w1, w1b, HID2 * HID);
    cast_f32_to_bf16<<<256,  256, 0, stream>>>(w2, w2b, NOUT * HID2);

    dim3 g1(HID2 / 128, TOKENS / 128);   // (32, 64)
    gemm1_gelu_kernel<<<g1, 256, 0, stream>>>(xb, w1b, b1, hb);

    router_kernel<<<TOKENS / 16, 32, 0, stream>>>(hb, w2b, b2, log_temp,
                                                  eps_logit, probs);

    const int total4 = TOKENS * NHEAD * HD / 4;
    combine_kernel<<<(total4 + 255) / 256, 256, 0, stream>>>(paths, probs, out);
}
